// Model_18614388261612
// MI455X (gfx1250) — matrix operand order repaired and checked by reading, not compiled, not run on hardware
//
#include <hip/hip_runtime.h>
#include <math.h>

// ---------------------------------------------------------------------------
// CDNA5 (gfx1250) Faster-RCNN/FPN inference.
// Conv + FC layers: v_wmma_f32_16x16x32_f16 implicit GEMM, f32 accumulate.
// Block = 8 waves = 128(M) x 16(N) macro-tile. The 32x16 B tile is gathered
// cooperatively once per K-step into double-buffered LDS (f16, transposed,
// padded stride -> ds_load_b128, no bank conflicts), shared by all 8 waves.
// A is fetched per-wave as contiguous vectorizable loads. K-loop main body is
// branch-free (clamped addresses + cndmask zeroing, magic-divide im2col).
// ---------------------------------------------------------------------------

typedef __attribute__((ext_vector_type(16))) _Float16 v16h;
typedef __attribute__((ext_vector_type(8)))  float    v8f;

#define NMS_IOU_TH 0.7f

// exact q = k / d for k < 2^24, d <= 2^9:  q = (k * magic33(d)) >> 33
__device__ __forceinline__ unsigned mdiv33(int k, unsigned long long magic)
{
    return (unsigned)(((unsigned long long)(unsigned)k * magic) >> 33);
}

// ========================= WMMA implicit-GEMM conv =========================
// out[o, oy, ox] = relu( sum_{c,ky,kx} in[c, iy, ix] * w[o, c, ky, kx] + b[o] )
// GEMM view: M = O (weight rows), N = OH*OW (im2col cols), K = C*KH*KW.
// Grid: tilesN * ceil(tilesM/8) blocks; block bm,bn -> waves cover
// tileM = bm*8+wid (clamped), shared tileN = bn.
__global__ __launch_bounds__(256)
void conv2d_wmma(const float* __restrict__ in, const float* __restrict__ w,
                 const float* __restrict__ bias, float* __restrict__ out,
                 int C, int H, int W, int O, int KH, int KW,
                 int stride, int pad, int OH, int OW, int relu,
                 unsigned long long magicKK, unsigned long long magicKW,
                 int tilesN)
{
    __shared__ _Float16 lbs[2][16][40];   // [buf][n][k], padded stride: 16B align

    const int tid  = threadIdx.x;
    const int lane = tid & 31;
    const int wid  = tid >> 5;
    const int half = lane >> 4;
    const int l15  = lane & 15;
    const int NP   = OH * OW;
    const int KK   = KH * KW;
    const int Ktot = C * KK;
    const int tilesM = (O + 15) >> 4;

    const int bn = blockIdx.x % tilesN;
    const int bm = blockIdx.x / tilesN;
    int tileM = bm * 8 + wid;
    if (tileM >= tilesM) tileM = tilesM - 1;   // clamp: keeps all waves at barriers
    const int tileN = bn;

    // ---- this thread's two staged B elements: e = 2*tid, 2*tid+1 ----
    // element e: kk = e>>4 (K offset within step), nn = e&15 (column).
    const int e0  = tid * 2;
    const int kk0 = e0 >> 4;
    const int nn0 = e0 & 15;          // even, so pair shares kk0
    int ncA = tileN * 16 + nn0;     if (ncA >= NP) ncA = NP - 1;
    int ncB = tileN * 16 + nn0 + 1; if (ncB >= NP) ncB = NP - 1;
    const int oyA = ncA / OW, oxA = ncA - oyA * OW;
    const int oyB = ncB / OW, oxB = ncB - oyB * OW;
    const int oysA = oyA * stride - pad, oxsA = oxA * stride - pad;
    const int oysB = oyB * stride - pad, oxsB = oxB * stride - pad;

    // per-lane output column
    const int  n      = tileN * 16 + l15;
    const bool nvalid = (n < NP);

    // A (weights) row for this lane (clamped; invalid rows masked at store)
    const int arow = tileM * 16 + l15;
    const float* wrow = w + (size_t)(arow < O ? arow : 0) * Ktot;

    v8f acc = {};
    const int Ksteps = (Ktot + 31) >> 5;

    for (int it = 0; it < Ksteps; ++it) {
        const int kb  = it << 5;
        const int buf = it & 1;

        // ---- cooperative im2col gather: 32x16 B tile -> LDS (f16, transposed)
        {
            int k  = kb + kk0;
            int kc = k < Ktot ? k : (Ktot - 1);
            int c  = (int)mdiv33(kc, magicKK);
            int r  = kc - c * KK;
            int ky = (int)mdiv33(r, magicKW);
            int kx = r - ky * KW;
            {
                int iy = oysA + ky, ix = oxsA + kx;
                bool inb = (k < Ktot) & (iy >= 0) & (iy < H) & (ix >= 0) & (ix < W);
                int iyc = iy < 0 ? 0 : (iy >= H ? H - 1 : iy);
                int ixc = ix < 0 ? 0 : (ix >= W ? W - 1 : ix);
                float v = in[((size_t)c * H + iyc) * W + ixc];   // unconditional
                lbs[buf][nn0][kk0] = (_Float16)(inb ? v : 0.0f);
            }
            {
                int iy = oysB + ky, ix = oxsB + kx;
                bool inb = (k < Ktot) & (iy >= 0) & (iy < H) & (ix >= 0) & (ix < W);
                int iyc = iy < 0 ? 0 : (iy >= H ? H - 1 : iy);
                int ixc = ix < 0 ? 0 : (ix >= W ? W - 1 : ix);
                float v = in[((size_t)c * H + iyc) * W + ixc];
                lbs[buf][nn0 + 1][kk0] = (_Float16)(inb ? v : 0.0f);
            }
        }
        __syncthreads();

        // ---- A fragment: e<8 -> k=kb+half*8+e ; e>=8 -> +16 ----
        v16h av, bv;
        if (kb + 32 <= Ktot) {
            const float* wp = wrow + kb + half * 8;
            __builtin_prefetch(wp + 32, 0, 1);
            #pragma unroll
            for (int e = 0; e < 8; ++e) av[e]     = (_Float16)wp[e];
            #pragma unroll
            for (int e = 0; e < 8; ++e) av[e + 8] = (_Float16)wp[e + 16];
        } else {   // ragged-K tail (conv1: Ktot = 147)
            #pragma unroll
            for (int e = 0; e < 16; ++e) {
                int k  = kb + ((e >> 3) * 16) + half * 8 + (e & 7);
                int kc = k < Ktot ? k : (Ktot - 1);
                float vA = wrow[kc];
                av[e] = (_Float16)(k < Ktot ? vA : 0.0f);
            }
        }
        // ---- B fragment from LDS: e<8 -> k=half*8+e ; e>=8 -> +16 (2 x ds_load_b128) ----
        const _Float16* lp = &lbs[buf][l15][half * 8];
        #pragma unroll
        for (int e = 0; e < 16; ++e) bv[e] = lp[(e >> 3) * 16 + (e & 7)];

        acc = __builtin_amdgcn_wmma_f32_16x16x32_f16(false, av, false, bv,
                                                     (short)0, acc, false, false);
    }

    // C/D layout: VGPR vv holds row m = vv + 8*half ; column = lane&15
    #pragma unroll
    for (int vv = 0; vv < 8; ++vv) {
        int m = tileM * 16 + vv + half * 8;
        if (m < O && nvalid) {
            float val = acc[vv] + bias[m];
            if (relu) val = fmaxf(val, 0.0f);
            out[(size_t)m * NP + n] = val;
        }
    }
}

// ============================ WMMA dense GEMM ==============================
// out[m, col_off + n] = (relu)( sum_k A[m,k]*B[k,n] + bias[n] ), row stride ldo
// Requirements (met by all call sites): K % 32 == 0, lda % 4 == 0, A 16B-aligned.
__global__ __launch_bounds__(256)
void gemm_wmma(const float* __restrict__ A, int lda,
               const float* __restrict__ B, int ldb,
               const float* __restrict__ bias,
               float* __restrict__ outp, int ldo, int col_off,
               int M, int N, int K, int relu, int tilesN)
{
    __shared__ _Float16 lbs[2][16][40];

    const int tid  = threadIdx.x;
    const int lane = tid & 31;
    const int wid  = tid >> 5;
    const int half = lane >> 4;
    const int l15  = lane & 15;
    const int tilesM = (M + 15) >> 4;

    const int bn = blockIdx.x % tilesN;
    const int bm = blockIdx.x / tilesN;
    int tileM = bm * 8 + wid;
    if (tileM >= tilesM) tileM = tilesM - 1;   // clamp: keeps all waves at barriers
    const int tileN = bn;

    const int e0  = tid * 2;
    const int kk0 = e0 >> 4;
    const int nn0 = e0 & 15;
    int ncA = tileN * 16 + nn0;     if (ncA >= N) ncA = N - 1;
    int ncB = tileN * 16 + nn0 + 1; if (ncB >= N) ncB = N - 1;

    const int nn = tileN * 16 + l15;
    const int ar = tileM * 16 + l15;
    const float* Arow = A + (size_t)(ar < M ? ar : 0) * lda;

    v8f acc = {};
    const int Ksteps = K >> 5;              // K % 32 == 0 at all call sites

    for (int it = 0; it < Ksteps; ++it) {
        const int kb  = it << 5;
        const int buf = it & 1;

        // ---- cooperative B stage: 32x16 tile -> LDS (f16, transposed) ----
        const float* brow = B + (size_t)(kb + kk0) * ldb;
        lbs[buf][nn0][kk0]     = (_Float16)brow[ncA];
        lbs[buf][nn0 + 1][kk0] = (_Float16)brow[ncB];
        __syncthreads();

        // ---- A fragment: four b128 loads ----
        const float4* ap = (const float4*)(Arow + kb + half * 8);
        __builtin_prefetch((const float*)ap + 32, 0, 1);
        float4 a0 = ap[0];
        float4 a1 = ap[1];
        float4 a2 = ap[4];      // +16 floats
        float4 a3 = ap[5];
        v16h av, bv;
        av[0]  = (_Float16)a0.x; av[1]  = (_Float16)a0.y;
        av[2]  = (_Float16)a0.z; av[3]  = (_Float16)a0.w;
        av[4]  = (_Float16)a1.x; av[5]  = (_Float16)a1.y;
        av[6]  = (_Float16)a1.z; av[7]  = (_Float16)a1.w;
        av[8]  = (_Float16)a2.x; av[9]  = (_Float16)a2.y;
        av[10] = (_Float16)a2.z; av[11] = (_Float16)a2.w;
        av[12] = (_Float16)a3.x; av[13] = (_Float16)a3.y;
        av[14] = (_Float16)a3.z; av[15] = (_Float16)a3.w;

        const _Float16* lp = &lbs[buf][l15][half * 8];
        #pragma unroll
        for (int e = 0; e < 16; ++e) bv[e] = lp[(e >> 3) * 16 + (e & 7)];

        acc = __builtin_amdgcn_wmma_f32_16x16x32_f16(false, av, false, bv,
                                                     (short)0, acc, false, false);
    }

    #pragma unroll
    for (int vv = 0; vv < 8; ++vv) {
        int m = tileM * 16 + vv + half * 8;
        if (m < M && nn < N) {
            float val = acc[vv] + bias[nn];
            if (relu) val = fmaxf(val, 0.0f);
            outp[(size_t)m * ldo + col_off + nn] = val;
        }
    }
}

// ============================== glue kernels ===============================
__global__ void maxpool3x3s2(const float* __restrict__ in, float* __restrict__ out,
                             int C, int H, int W, int OH, int OW)
{
    int idx = blockIdx.x * blockDim.x + threadIdx.x;
    int tot = C * OH * OW;
    if (idx >= tot) return;
    int ox = idx % OW;
    int t  = idx / OW;
    int oy = t % OH;
    int c  = t / OH;
    float m = -1e30f;
    for (int ky = 0; ky < 3; ++ky) {
        int iy = oy * 2 - 1 + ky;
        if (iy < 0 || iy >= H) continue;
        for (int kx = 0; kx < 3; ++kx) {
            int ix = ox * 2 - 1 + kx;
            if (ix < 0 || ix >= W) continue;
            m = fmaxf(m, in[((size_t)c * H + iy) * W + ix]);
        }
    }
    out[idx] = m;
}

// dst[c,y,x] += src[c, y/2, x/2]   (dst: C x H x W, src: C x H/2 x W/2)
__global__ void add_up2(float* __restrict__ dst, const float* __restrict__ src,
                        int C, int H, int W)
{
    int idx = blockIdx.x * blockDim.x + threadIdx.x;
    int tot = C * H * W;
    if (idx >= tot) return;
    int x = idx % W;
    int t = idx / W;
    int y = t % H;
    int c = t / H;
    dst[idx] += src[((size_t)c * (H / 2) + (y >> 1)) * (W / 2) + (x >> 1)];
}

// dst[c,y,x] = src[c, 2y, 2x]   (dst: C x H x W, src: C x 2H x 2W)
__global__ void subsample2(float* __restrict__ dst, const float* __restrict__ src,
                           int C, int H, int W)
{
    int idx = blockIdx.x * blockDim.x + threadIdx.x;
    int tot = C * H * W;
    if (idx >= tot) return;
    int x = idx % W;
    int t = idx / W;
    int y = t % H;
    int c = t / H;
    dst[idx] = src[((size_t)c * (2 * H) + 2 * y) * (2 * W) + 2 * x];
}

// Per-anchor: fg score (softmax[:,1]), box transform, clip -> props; (score,idx)
__global__ void proposal_prep(const float* __restrict__ obj, const float* __restrict__ reg,
                              float* __restrict__ props, float2* __restrict__ pairs,
                              int Hf, int Wf, float asize, int N, int padN)
{
    int i = blockIdx.x * blockDim.x + threadIdx.x;
    if (i >= padN) return;
    if (i >= N) { pairs[i] = make_float2(-1e30f, (float)i); return; }

    int a   = i % 3;
    int pos = i / 3;            // y*Wf + x
    int x   = pos % Wf;
    int y   = pos / Wf;
    int HW  = Hf * Wf;

    float o0 = obj[(size_t)(2 * a)     * HW + pos];
    float o1 = obj[(size_t)(2 * a + 1) * HW + pos];
    float fg = 1.0f / (1.0f + expf(o0 - o1));   // softmax([o0,o1])[1]

    float t0 = reg[(size_t)(4 * a + 0) * HW + pos];
    float t1 = reg[(size_t)(4 * a + 1) * HW + pos];
    float t2 = reg[(size_t)(4 * a + 2) * HW + pos];
    float t3 = reg[(size_t)(4 * a + 3) * HW + pos];

    const float ratios[3] = {0.5f, 1.0f, 2.0f};
    float sr = sqrtf(ratios[a]);
    float aw = asize * sr;
    float ah = asize / sr;
    float cx = ((float)x + 0.5f) * (512.0f / (float)Wf);
    float cy = ((float)y + 0.5f) * (512.0f / (float)Hf);

    float ncx = t0 * aw + cx;
    float ncy = t1 * ah + cy;
    float nw  = expf(t2) * aw;
    float nh  = expf(t3) * ah;

    float x1 = fminf(fmaxf(ncx - 0.5f * nw, 0.0f), 512.0f);
    float y1 = fminf(fmaxf(ncy - 0.5f * nh, 0.0f), 512.0f);
    float x2 = fminf(fmaxf(ncx + 0.5f * nw, 0.0f), 512.0f);
    float y2 = fminf(fmaxf(ncy + 0.5f * nh, 0.0f), 512.0f);

    props[4 * i + 0] = x1;
    props[4 * i + 1] = y1;
    props[4 * i + 2] = x2;
    props[4 * i + 3] = y2;
    pairs[i] = make_float2(fg, (float)i);
}

__device__ __forceinline__ bool pair_before(float2 a, float2 b)
{   // descending score, ties -> lower original index first (lax.top_k semantics)
    return (a.x > b.x) || (a.x == b.x && a.y < b.y);
}

__global__ void bitonic_step(float2* __restrict__ p, int n, int k, int j)
{
    int i = blockIdx.x * blockDim.x + threadIdx.x;
    if (i >= n) return;
    int ixj = i ^ j;
    if (ixj <= i || ixj >= n) return;
    float2 pi = p[i], pj = p[ixj];
    bool up = ((i & k) == 0);                   // "better-first" block direction
    bool sw = up ? pair_before(pj, pi) : pair_before(pi, pj);
    if (sw) { p[i] = pj; p[ixj] = pi; }
}

__global__ void gather_sorted(const float2* __restrict__ pairs,
                              const float* __restrict__ props,
                              float* __restrict__ selboxes, int pre)
{
    int i = blockIdx.x * blockDim.x + threadIdx.x;
    if (i >= pre) return;
    int idx = (int)pairs[i].y;
    selboxes[4 * i + 0] = props[4 * idx + 0];
    selboxes[4 * i + 1] = props[4 * idx + 1];
    selboxes[4 * i + 2] = props[4 * idx + 2];
    selboxes[4 * i + 3] = props[4 * idx + 3];
}

// Sequential greedy NMS on score-sorted boxes, then keep first `post`
// unsuppressed (falling back to suppressed in index order = top_k on -inf'd sel)
__global__ __launch_bounds__(256)
void nms_kernel(const float* __restrict__ boxes, int n, int post,
                float* __restrict__ outb)
{
    __shared__ float bx[2000][4];
    __shared__ float area[2000];
    __shared__ unsigned char sup[2000];

    for (int i = threadIdx.x; i < n; i += blockDim.x) {
        float x1 = boxes[4 * i + 0], y1 = boxes[4 * i + 1];
        float x2 = boxes[4 * i + 2], y2 = boxes[4 * i + 3];
        bx[i][0] = x1; bx[i][1] = y1; bx[i][2] = x2; bx[i][3] = y2;
        area[i] = (x2 - x1) * (y2 - y1);
        sup[i] = 0;
    }
    __syncthreads();

    for (int i = 0; i < n - 1; ++i) {
        if (!sup[i]) {
            float ax1 = bx[i][0], ay1 = bx[i][1], ax2 = bx[i][2], ay2 = bx[i][3];
            float aa = area[i];
            for (int j = i + 1 + (int)threadIdx.x; j < n; j += blockDim.x) {
                float lx = fmaxf(ax1, bx[j][0]);
                float ly = fmaxf(ay1, bx[j][1]);
                float rx = fminf(ax2, bx[j][2]);
                float ry = fminf(ay2, bx[j][3]);
                float iw = fmaxf(rx - lx, 0.0f);
                float ih = fmaxf(ry - ly, 0.0f);
                float inter = iw * ih;
                float iou = inter / (aa + area[j] - inter + 1e-9f);
                if (iou > NMS_IOU_TH) sup[j] = 1;
            }
        }
        __syncthreads();
    }

    if (threadIdx.x == 0) {
        int cnt = 0;
        for (int i = 0; i < n && cnt < post; ++i)
            if (!sup[i]) {
                outb[4 * cnt + 0] = bx[i][0]; outb[4 * cnt + 1] = bx[i][1];
                outb[4 * cnt + 2] = bx[i][2]; outb[4 * cnt + 3] = bx[i][3];
                ++cnt;
            }
        for (int i = 0; i < n && cnt < post; ++i)
            if (sup[i]) {
                outb[4 * cnt + 0] = bx[i][0]; outb[4 * cnt + 1] = bx[i][1];
                outb[4 * cnt + 2] = bx[i][2]; outb[4 * cnt + 3] = bx[i][3];
                ++cnt;
            }
    }
}

__global__ void kassign(const float* __restrict__ boxes, int* __restrict__ kbuf, int R)
{
    int r = blockIdx.x * blockDim.x + threadIdx.x;
    if (r >= R) return;
    float w = fmaxf(boxes[4 * r + 2] - boxes[4 * r + 0], 0.001f);
    float h = fmaxf(boxes[4 * r + 3] - boxes[4 * r + 1], 0.001f);
    float k = floorf(4.0f + log2f(sqrtf(w * h) / 224.0f));
    k = fminf(fmaxf(k, 2.0f), 5.0f);
    kbuf[r] = (int)k - 2;     // 0..3 -> p2..p5
}

__global__ __launch_bounds__(64)
void roi_align_kernel(const float* __restrict__ p2, const float* __restrict__ p3,
                      const float* __restrict__ p4, const float* __restrict__ p5,
                      const float* __restrict__ boxes, const int* __restrict__ kbuf,
                      float* __restrict__ pool, int R)
{
    int r = blockIdx.x;
    int c = blockIdx.y;
    int t = threadIdx.x;
    if (r >= R || t >= 49) return;
    int py = t / 7, px = t % 7;

    int lvl = kbuf[r];
    const float* feat;
    int Hf; float stride;
    if      (lvl == 0) { feat = p2; Hf = 128; stride = 4.0f;  }
    else if (lvl == 1) { feat = p3; Hf = 64;  stride = 8.0f;  }
    else if (lvl == 2) { feat = p4; Hf = 32;  stride = 16.0f; }
    else               { feat = p5; Hf = 16;  stride = 32.0f; }
    int Wf = Hf;

    float x1 = boxes[4 * r + 0], y1 = boxes[4 * r + 1];
    float x2 = boxes[4 * r + 2], y2 = boxes[4 * r + 3];
    float bw = fmaxf(x2 - x1, 1.0f);
    float bh = fmaxf(y2 - y1, 1.0f);

    float ys = (y1 + ((float)py + 0.5f) * bh / 7.0f) / stride - 0.5f;
    float xs = (x1 + ((float)px + 0.5f) * bw / 7.0f) / stride - 0.5f;
    ys = fminf(fmaxf(ys, 0.0f), (float)Hf - 1.0f);
    xs = fminf(fmaxf(xs, 0.0f), (float)Wf - 1.0f);

    int y0 = (int)floorf(ys); y0 = y0 < 0 ? 0 : (y0 > Hf - 2 ? Hf - 2 : y0);
    int x0 = (int)floorf(xs); x0 = x0 < 0 ? 0 : (x0 > Wf - 2 ? Wf - 2 : x0);
    float dy = ys - (float)y0;
    float dx = xs - (float)x0;

    const float* f = feat + (size_t)c * Hf * Wf;
    float f00 = f[y0 * Wf + x0];
    float f01 = f[y0 * Wf + x0 + 1];
    float f10 = f[(y0 + 1) * Wf + x0];
    float f11 = f[(y0 + 1) * Wf + x0 + 1];
    float val = f00 * (1 - dy) * (1 - dx) + f01 * (1 - dy) * dx
              + f10 * dy * (1 - dx)       + f11 * dy * dx;

    pool[((size_t)r * 256 + c) * 49 + py * 7 + px] = val;
}

// ============================ host-side helpers ============================
static inline unsigned long long magic33(int d)
{
    return ((1ull << 33) + (unsigned long long)d - 1) / (unsigned long long)d;
}

static void conv(hipStream_t s, const float* in, const float* w, const float* b,
                 float* out, int C, int H, int W, int O, int KH, int KW,
                 int stride, int pad, int relu)
{
    int OH = (H + 2 * pad - KH) / stride + 1;
    int OW = (W + 2 * pad - KW) / stride + 1;
    int tilesN = (OH * OW + 15) / 16;
    int tilesM = (O + 15) / 16;
    int mblk   = (tilesM + 7) / 8;
    conv2d_wmma<<<tilesN * mblk, 256, 0, s>>>(in, w, b, out, C, H, W, O, KH, KW,
                                              stride, pad, OH, OW, relu,
                                              magic33(KH * KW), magic33(KW), tilesN);
}

static void gemm(hipStream_t s, const float* A, int lda, const float* B, int ldb,
                 const float* bias, float* outp, int ldo, int col_off,
                 int M, int N, int K, int relu)
{
    int tilesN = (N + 15) / 16;
    int tilesM = (M + 15) / 16;
    int mblk   = (tilesM + 7) / 8;
    gemm_wmma<<<tilesN * mblk, 256, 0, s>>>(A, lda, B, ldb, bias, outp, ldo, col_off,
                                            M, N, K, relu, tilesN);
}

static void sort_pairs(hipStream_t s, float2* pairs, int padN)
{
    int blocks = (padN + 255) / 256;
    for (int k = 2; k <= padN; k <<= 1)
        for (int j = k >> 1; j > 0; j >>= 1)
            bitonic_step<<<blocks, 256, 0, s>>>(pairs, padN, k, j);
}

// ================================ driver ===================================
extern "C" void kernel_launch(void* const* d_in, const int* in_sizes, int n_in,
                              void* d_out, int out_size, void* d_ws, size_t ws_size,
                              hipStream_t stream)
{
    (void)in_sizes; (void)n_in; (void)out_size; (void)ws_size;

    const float* image     = (const float*)d_in[0];
    const float* conv1_w   = (const float*)d_in[1];
    const float* conv1_b   = (const float*)d_in[2];
    const float* conv2_w   = (const float*)d_in[3];
    const float* conv2_b   = (const float*)d_in[4];
    const float* conv3_w   = (const float*)d_in[5];
    const float* conv3_b   = (const float*)d_in[6];
    const float* conv4_w   = (const float*)d_in[7];
    const float* conv4_b   = (const float*)d_in[8];
    const float* conv5_w   = (const float*)d_in[9];
    const float* conv5_b   = (const float*)d_in[10];
    const float* lat2_w    = (const float*)d_in[11];
    const float* lat2_b    = (const float*)d_in[12];
    const float* lat3_w    = (const float*)d_in[13];
    const float* lat3_b    = (const float*)d_in[14];
    const float* lat4_w    = (const float*)d_in[15];
    const float* lat4_b    = (const float*)d_in[16];
    const float* lat5_w    = (const float*)d_in[17];
    const float* lat5_b    = (const float*)d_in[18];
    const float* deal2_w   = (const float*)d_in[19];
    const float* deal2_b   = (const float*)d_in[20];
    const float* deal3_w   = (const float*)d_in[21];
    const float* deal3_b   = (const float*)d_in[22];
    const float* deal4_w   = (const float*)d_in[23];
    const float* deal4_b   = (const float*)d_in[24];
    const float* rpnc_w    = (const float*)d_in[25];
    const float* rpnc_b    = (const float*)d_in[26];
    const float* rpno_w    = (const float*)d_in[27];
    const float* rpno_b    = (const float*)d_in[28];
    const float* rpnr_w    = (const float*)d_in[29];
    const float* rpnr_b    = (const float*)d_in[30];
    const float* fc1_w     = (const float*)d_in[31];
    const float* fc1_b     = (const float*)d_in[32];
    const float* fc2_w     = (const float*)d_in[33];
    const float* fc2_b     = (const float*)d_in[34];
    const float* cls_w     = (const float*)d_in[35];
    const float* cls_b     = (const float*)d_in[36];
    const float* reg_w     = (const float*)d_in[37];
    const float* reg_b     = (const float*)d_in[38];

    float* out = (float*)d_out;   // (992, 105) row-major

    // ------------------ workspace layout (floats, 1KB aligned) ---------------
    float* ws = (float*)d_ws;
    size_t off = 0;
    auto alloc = [&](size_t nf) -> float* {
        float* p = ws + off;
        off += (nf + 255) & ~(size_t)255;
        return p;
    };

    float* c1    = alloc((size_t)64 * 256 * 256);
    float* pool1 = alloc((size_t)64 * 128 * 128);
    float* c2    = alloc((size_t)256 * 128 * 128);
    float* c3    = alloc((size_t)512 * 64 * 64);
    float* c4    = alloc((size_t)1024 * 32 * 32);
    float* c5    = alloc((size_t)2048 * 16 * 16);
    float* p5    = alloc((size_t)256 * 16 * 16);
    float* p4pre = alloc((size_t)256 * 32 * 32);
    float* p3pre = alloc((size_t)256 * 64 * 64);
    float* p2pre = alloc((size_t)256 * 128 * 128);
    float* p4d   = alloc((size_t)256 * 32 * 32);
    float* p3d   = alloc((size_t)256 * 64 * 64);
    float* p2d   = alloc((size_t)256 * 128 * 128);
    float* p6    = alloc((size_t)256 * 8 * 8);
    float* prop_all = alloc((size_t)992 * 4);
    int*   kbuf  = (int*)alloc(992);

    size_t phase = off;
    // Phase A (RPN / proposals), aliased with Phase B (ROI pool / FC heads)
    float*  rpnh  = alloc((size_t)512 * 128 * 128);
    float*  objb  = alloc((size_t)6 * 128 * 128);
    float*  regb  = alloc((size_t)12 * 128 * 128);
    float*  props = alloc((size_t)49152 * 4);
    float2* pairs = (float2*)alloc((size_t)65536 * 2);
    float*  selb  = alloc((size_t)2000 * 4);

    off = phase;   // reuse same region (temporally disjoint in stream order)
    float* pool = alloc((size_t)992 * 256 * 49);
    float* h1   = alloc((size_t)992 * 1024);
    float* h2   = alloc((size_t)992 * 1024);

    // ------------------------------ backbone ------------------------------
    conv(stream, image, conv1_w, conv1_b, c1,   3, 512, 512,   64, 7, 7, 2, 3, 1);
    {
        int tot = 64 * 128 * 128;
        maxpool3x3s2<<<(tot + 255) / 256, 256, 0, stream>>>(c1, pool1, 64, 256, 256, 128, 128);
    }
    conv(stream, pool1, conv2_w, conv2_b, c2,   64, 128, 128,  256, 3, 3, 1, 1, 1);
    conv(stream, c2,    conv3_w, conv3_b, c3,  256, 128, 128,  512, 3, 3, 2, 1, 1);
    conv(stream, c3,    conv4_w, conv4_b, c4,  512,  64,  64, 1024, 3, 3, 2, 1, 1);
    conv(stream, c4,    conv5_w, conv5_b, c5, 1024,  32,  32, 2048, 3, 3, 2, 1, 1);

    // -------------------------------- FPN ---------------------------------
    conv(stream, c5, lat5_w, lat5_b, p5,    2048, 16, 16, 256, 1, 1, 1, 0, 0);
    conv(stream, c4, lat4_w, lat4_b, p4pre, 1024, 32, 32, 256, 1, 1, 1, 0, 0);
    {
        int tot = 256 * 32 * 32;
        add_up2<<<(tot + 255) / 256, 256, 0, stream>>>(p4pre, p5, 256, 32, 32);
    }
    conv(stream, c3, lat3_w, lat3_b, p3pre, 512, 64, 64, 256, 1, 1, 1, 0, 0);
    {
        int tot = 256 * 64 * 64;
        add_up2<<<(tot + 255) / 256, 256, 0, stream>>>(p3pre, p4pre, 256, 64, 64);
    }
    conv(stream, c2, lat2_w, lat2_b, p2pre, 256, 128, 128, 256, 1, 1, 1, 0, 0);
    {
        int tot = 256 * 128 * 128;
        add_up2<<<(tot + 255) / 256, 256, 0, stream>>>(p2pre, p3pre, 256, 128, 128);
    }
    conv(stream, p4pre, deal4_w, deal4_b, p4d, 256, 32, 32,   256, 3, 3, 1, 1, 0);
    conv(stream, p3pre, deal3_w, deal3_b, p3d, 256, 64, 64,   256, 3, 3, 1, 1, 0);
    conv(stream, p2pre, deal2_w, deal2_b, p2d, 256, 128, 128, 256, 3, 3, 1, 1, 0);
    {
        int tot = 256 * 8 * 8;
        subsample2<<<(tot + 255) / 256, 256, 0, stream>>>(p6, p5, 256, 8, 8);
    }

    // -------------------------- RPN + proposals ---------------------------
    struct Lvl { const float* feat; int Hf; float asize; };
    Lvl lv[5] = { {p2d, 128, 32.0f}, {p3d, 64, 64.0f}, {p4d, 32, 128.0f},
                  {p5, 16, 256.0f}, {p6, 8, 512.0f} };

    int prop_off = 0;
    for (int L = 0; L < 5; ++L) {
        int Hf = lv[L].Hf, Wf = Hf;
        int Nanch = Hf * Wf * 3;
        int pre  = Nanch < 2000 ? Nanch : 2000;
        int post = pre < 200 ? pre : 200;
        int padN = 1; while (padN < Nanch) padN <<= 1;

        conv(stream, lv[L].feat, rpnc_w, rpnc_b, rpnh, 256, Hf, Wf, 512, 3, 3, 1, 1, 1);
        conv(stream, rpnh, rpno_w, rpno_b, objb, 512, Hf, Wf,  6, 1, 1, 1, 0, 0);
        conv(stream, rpnh, rpnr_w, rpnr_b, regb, 512, Hf, Wf, 12, 1, 1, 1, 0, 0);

        proposal_prep<<<(padN + 255) / 256, 256, 0, stream>>>(
            objb, regb, props, pairs, Hf, Wf, lv[L].asize, Nanch, padN);
        sort_pairs(stream, pairs, padN);
        gather_sorted<<<(pre + 255) / 256, 256, 0, stream>>>(pairs, props, selb, pre);
        nms_kernel<<<1, 256, 0, stream>>>(selb, pre, post, prop_all + 4 * prop_off);
        prop_off += post;
    }
    const int R = prop_off;   // 200*4 + 192 = 992

    // ---------------------------- ROI align + FC --------------------------
    kassign<<<(R + 255) / 256, 256, 0, stream>>>(prop_all, kbuf, R);
    {
        dim3 grid(R, 256);
        roi_align_kernel<<<grid, 64, 0, stream>>>(p2d, p3d, p4d, p5,
                                                  prop_all, kbuf, pool, R);
    }
    gemm(stream, pool, 12544, fc1_w, 1024, fc1_b, h1, 1024, 0, R, 1024, 12544, 1);
    gemm(stream, h1,   1024,  fc2_w, 1024, fc2_b, h2, 1024, 0, R, 1024, 1024,  1);
    gemm(stream, h2,   1024,  cls_w,   21, cls_b, out, 105, 0,  R,   21, 1024, 0);
    gemm(stream, h2,   1024,  reg_w,   84, reg_b, out, 105, 21, R,   84, 1024, 0);
}
